// DiverseRegDCConv2d_20538533609774
// MI455X (gfx1250) — matrix operand order repaired and checked by reading, not compiled, not run on hardware
//
#include <hip/hip_runtime.h>
#include <hip/hip_bf16.h>

// ---------------------------------------------------------------------------
// DiverseRegDCConv2d on MI455X (gfx1250, wave32, WMMA + async-to-LDS).
//
// Conv = 32 implicit GEMMs (M=256, N=784, K=2304) = 29.6 GFLOP, run on
// v_wmma_f32_16x16x32_bf16 with a bf16 hi/lo split (3 WMMAs / tile-pair,
// ~1e-5 rel err vs f32).  Memory floor ~230 MB / 23.3 TB/s ~ 10 us; the
// 18.9 MB weight bank and all generated panels are L2-resident (192 MB).
//
// Inner loop: double-buffered B panel (im2col) filled with
// GLOBAL_LOAD_ASYNC_TO_LDS_B128 (per-lane global->LDS, ASYNCcnt), overlapped
// with the 12 WMMAs of the previous chunk; sync = s_wait_asynccnt + barrier.
//
// ws layout (requires ws_size >= 101,187,840 bytes):
//   [ 0)            Whi : bf16 [B][O][K=2304]      (K order = r*256 + c)
//   [37,748,736)    Wlo : bf16 [B][O][K=2304]
//   [75,497,472)    Xhi : bf16 [B][784][C=256]     (channels contiguous)
//   [88,342,528)    Xlo : bf16 [B][784][C=256]
//   [101,187,584)   zero page (256 B, memset)      (padding redirect)
// ---------------------------------------------------------------------------

#define B_    32
#define CIN   256
#define COUT  256
#define H_    28
#define W_    28
#define NUM_  8
#define KDIM  (CIN * 9)           // 2304
#define NSP   (H_ * W_)           // 784

#define WHI_OFF  0ll
#define WLO_OFF  37748736ll
#define XHI_OFF  75497472ll
#define XLO_OFF  88342528ll
#define ZP_OFF   101187584ll
#define WS_NEED  (ZP_OFF + 256)

typedef __attribute__((ext_vector_type(16))) __bf16 v16bf;
typedef __attribute__((ext_vector_type(8)))  float  v8f;

union FragAB { v16bf v; uint4 u[2]; };

// Per-lane async global->LDS copy (GVS mode: saddr64 + vgpr32 offset).
// vdst = wave-relative LDS byte address, tracked by ASYNCcnt.
__device__ __forceinline__ void async_g2l_b128(unsigned lds_byte, int goff_byte,
                                               const void* sbase) {
    asm volatile("global_load_async_to_lds_b128 %0, %1, %2"
                 :: "v"(lds_byte), "v"(goff_byte), "s"(sbase)
                 : "memory");
}
__device__ __forceinline__ void wait_async0() {
    asm volatile("s_wait_asynccnt 0x0" ::: "memory");
}
// Generic (flat) pointer to a __shared__ object: addr[31:0] IS the
// wave-relative LDS byte address (ISA flat->LDS aperture mapping).
__device__ __forceinline__ unsigned lds_byte_addr(const void* p) {
    return (unsigned)(unsigned long long)p;
}

// ---------------------------------------------------------------------------
// Kernel 1: per-sample filter generation + bf16 hi/lo split.
//   wgen[b][o][kk] = sum_n se[b][n] * bank[((o*CIN + c)*9 + r)*8 + n],
//   kk = r*256 + c (kernel-position-major K).  Bank stays L2-resident.
// ---------------------------------------------------------------------------
__global__ __launch_bounds__(256)
void dcgen_kernel(const float* __restrict__ se, const float* __restrict__ bank,
                  __bf16* __restrict__ whi, __bf16* __restrict__ wlo) {
    long idx = (long)blockIdx.x * 256 + threadIdx.x;     // over B*O*KDIM
    int  kk  = (int)(idx % KDIM);
    long t   = idx / KDIM;
    int  o   = (int)(t % COUT);
    int  b   = (int)(t / COUT);
    int  r   = kk >> 8;
    int  c   = kk & 255;

    long wrow = (((long)o * CIN + c) * 9 + r) * NUM_;    // 32B-aligned
    const float4* wp = (const float4*)(bank + wrow);
    float4 w0 = wp[0], w1 = wp[1];
    const float* sp = se + b * NUM_;
    float acc = w0.x * sp[0] + w0.y * sp[1] + w0.z * sp[2] + w0.w * sp[3]
              + w1.x * sp[4] + w1.y * sp[5] + w1.z * sp[6] + w1.w * sp[7];

    __bf16 hi = (__bf16)acc;
    __bf16 lo = (__bf16)(acc - (float)hi);
    long oidx = ((long)b * COUT + o) * KDIM + kk;
    whi[oidx] = hi;
    wlo[oidx] = lo;
}

// ---------------------------------------------------------------------------
// Kernel 1b: input bf16 hi/lo split, transposed to [b][spatial][c] so the
// conv kernel's im2col rows are 64 B contiguous (async b128 friendly).
// ---------------------------------------------------------------------------
__global__ __launch_bounds__(256)
void xsplit_kernel(const float* __restrict__ x,
                   __bf16* __restrict__ xhi, __bf16* __restrict__ xlo) {
    long idx = (long)blockIdx.x * 256 + threadIdx.x;     // over B*NSP*CIN
    int  c   = (int)(idx % CIN);
    long t   = idx / CIN;
    int  n   = (int)(t % NSP);
    int  b   = (int)(t / NSP);
    float v  = x[((long)b * CIN + c) * NSP + n];
    __bf16 hi = (__bf16)v;
    __bf16 lo = (__bf16)(v - (float)hi);
    xhi[idx] = hi;                                       // [b][n][c] order
    xlo[idx] = lo;
}

// ---------------------------------------------------------------------------
// Kernel 2: pipelined implicit-GEMM conv.
//   Workgroup (256 thr = 8 waves): tile M=128, N=64, K chunked by 32.
//   Wave w owns rows [m_base + 16w, +16) x four 16x16 N-subtiles.
//   B panel double-buffered in LDS, filled by async-to-LDS b128 (2 per
//   thread per chunk), overlapped with WMMA of the current chunk.
// ---------------------------------------------------------------------------
__global__ __launch_bounds__(256)
void dcconv_wmma_kernel(const char* __restrict__ ws,
                        const float* __restrict__ bias,
                        float* __restrict__ out) {
    __shared__ __bf16 Bhi[2][64][32];   // [buf][n_local][k_local], 2 x 4 KB
    __shared__ __bf16 Blo[2][64][32];

    const int tid  = threadIdx.x;
    const int lane = tid & 31;
    const int wid  = tid >> 5;
    const int half = lane >> 4;
    const int lrow = lane & 15;

    const int b      = blockIdx.z;
    const int m_base = blockIdx.y * 128 + wid * 16;
    const int n_base = blockIdx.x * 64;

    // Fill mapping: thread owns one 16 B segment: row n_loc, k [8*seg, 8*seg+8).
    const int  n_loc  = tid >> 2;                 // 0..63
    const int  seg    = tid & 3;                  // 0..3
    const int  n_g    = n_base + n_loc;
    const int  py     = n_g / W_;
    const int  px     = n_g % W_;
    const bool nvalid = (n_g < NSP);
    const int  bNSP   = b * NSP;

    const unsigned ldsH0 = lds_byte_addr(&Bhi[0][n_loc][seg * 8]);
    const unsigned ldsL0 = lds_byte_addr(&Blo[0][n_loc][seg * 8]);
    const int      segB  = seg * 16;

    const __bf16* whi = (const __bf16*)(ws + WHI_OFF);
    const __bf16* wlo = (const __bf16*)(ws + WLO_OFF);

    // Issue async fill for chunk `kc` into buffer `buf`.
    auto fill = [&](int kc, int buf) {
        const int r  = kc >> 8;                   // constant within chunk
        const int ky = r / 3, kx = r % 3;
        const int cb = kc & 255;
        const int iy = py + ky - 1;
        const int ix = px + kx - 1;
        const bool ok = nvalid & (iy >= 0) & (iy < H_) & (ix >= 0) & (ix < W_);
        int offH, offL;
        if (ok) {
            int sp = bNSP + iy * W_ + ix;
            offH = (int)XHI_OFF + (sp * CIN + cb) * 2 + segB;
            offL = offH + (int)(XLO_OFF - XHI_OFF);
        } else {
            offH = (int)ZP_OFF;
            offL = (int)ZP_OFF;
        }
        const unsigned bufB = (unsigned)buf * (64 * 32 * 2);
        async_g2l_b128(ldsH0 + bufB, offH, ws);
        async_g2l_b128(ldsL0 + bufB, offL, ws);
    };

    v8f acc[4] = {};

    fill(0, 0);
    for (int i = 0; i < KDIM / 32; ++i) {
        const int p  = i & 1;
        const int kc = i * 32;
        wait_async0();          // my chunk-i transfers landed in LDS
        __syncthreads();        // everyone's landed; prev reads of buf 1-p done
        if (i + 1 < KDIM / 32) fill(kc + 32, 1 - p);   // overlap with WMMAs

        // A fragments (lanes<16 -> K 0..15,
        // lanes>=16 -> K 16..31; row M = lane%16). 16B-aligned.
        const long arow = ((long)b * COUT + m_base + lrow) * KDIM + kc + half * 16;
        FragAB ahi, alo;
        ahi.u[0] = *(const uint4*)(whi + arow);
        ahi.u[1] = *(const uint4*)(whi + arow + 8);
        alo.u[0] = *(const uint4*)(wlo + arow);
        alo.u[1] = *(const uint4*)(wlo + arow + 8);

        #pragma unroll
        for (int ns = 0; ns < 4; ++ns) {
            // B fragment: lanes<16 -> col N=lane, K 0..15; lanes>=16 -> K 16..31.
            const __bf16* bph = &Bhi[p][ns * 16 + lrow][half * 16];
            const __bf16* bpl = &Blo[p][ns * 16 + lrow][half * 16];
            FragAB bhi, blo;
            bhi.u[0] = *(const uint4*)(bph);
            bhi.u[1] = *(const uint4*)(bph + 8);
            blo.u[0] = *(const uint4*)(bpl);
            blo.u[1] = *(const uint4*)(bpl + 8);

            // bf16x3: hi*hi + hi*lo + lo*hi
            acc[ns] = __builtin_amdgcn_wmma_f32_16x16x32_bf16(
                false, ahi.v, false, bhi.v, (short)0, acc[ns], false, false);
            acc[ns] = __builtin_amdgcn_wmma_f32_16x16x32_bf16(
                false, ahi.v, false, blo.v, (short)0, acc[ns], false, false);
            acc[ns] = __builtin_amdgcn_wmma_f32_16x16x32_bf16(
                false, alo.v, false, bhi.v, (short)0, acc[ns], false, false);
        }
    }

    // Epilogue: C/D layout -> N = lane&15, M = (lane>>4)*8 + j.
    const int mrow0 = m_base + half * 8;
    #pragma unroll
    for (int ns = 0; ns < 4; ++ns) {
        int n_out = n_base + ns * 16 + lrow;
        if (n_out < NSP) {
            #pragma unroll
            for (int j = 0; j < 8; ++j) {
                int o = mrow0 + j;
                out[((long)b * COUT + o) * NSP + n_out] = acc[ns][j] + bias[o];
            }
        }
    }
}

// ---------------------------------------------------------------------------
extern "C" void kernel_launch(void* const* d_in, const int* in_sizes, int n_in,
                              void* d_out, int out_size, void* d_ws, size_t ws_size,
                              hipStream_t stream) {
    const float* x    = (const float*)d_in[0];   // [32,256,28,28]
    const float* se   = (const float*)d_in[1];   // [32,8]
    const float* bank = (const float*)d_in[2];   // [256*256*9, 8]
    const float* bias = (const float*)d_in[3];   // [256]
    float* out = (float*)d_out;                  // [32,256,28,28]
    char*  ws  = (char*)d_ws;                    // needs WS_NEED bytes

    (void)hipMemsetAsync(ws + ZP_OFF, 0, 256, stream);   // zero pad page

    const long wtotal = (long)B_ * COUT * KDIM;          // 18,874,368
    dcgen_kernel<<<dim3((unsigned)(wtotal / 256)), 256, 0, stream>>>(
        se, bank, (__bf16*)(ws + WHI_OFF), (__bf16*)(ws + WLO_OFF));

    const long xtotal = (long)B_ * NSP * CIN;            // 6,422,528
    xsplit_kernel<<<dim3((unsigned)(xtotal / 256)), 256, 0, stream>>>(
        x, (__bf16*)(ws + XHI_OFF), (__bf16*)(ws + XLO_OFF));

    dim3 grid((NSP + 63) / 64, COUT / 128, B_);          // (13, 2, 32)
    dcconv_wmma_kernel<<<grid, 256, 0, stream>>>(ws, bias, out);
}
